// ProbSparseSelfAttention_58574763983211
// MI455X (gfx1250) — matrix operand order repaired and checked by reading, not compiled, not run on hardware
//
#include <hip/hip_runtime.h>
#include <hip/hip_bf16.h>

// ---------------------------------------------------------------------------
// ProbSparse self-attention for MI455X (gfx1250), bf16 WMMA everywhere.
// B=2, L=2048, d_model=1024, H=16, Dh=64, U = int(5*ln(2048)) = 38.
// ---------------------------------------------------------------------------

typedef __bf16 bf16;
typedef __attribute__((ext_vector_type(16))) __bf16 v16bf;
typedef __attribute__((ext_vector_type(8)))  __bf16 v8bf;
typedef __attribute__((ext_vector_type(8)))  float  v8f;

static constexpr int B_ = 2;
static constexpr int L_ = 2048;
static constexpr int DM = 1024;
static constexpr int H_ = 16;
static constexpr int DH = 64;
static constexpr int U_ = 38;

#define WMMA_BF16(a, b, c) \
  __builtin_amdgcn_wmma_f32_16x16x32_bf16(false, (a), false, (b), (short)0, (c), false, false)

__device__ inline v8f zero8() { v8f z = {0.f,0.f,0.f,0.f,0.f,0.f,0.f,0.f}; return z; }

// float -> bf16 via native fptrunc (backend picks hw conversion on gfx1250)
__device__ inline bf16 f2bf(float f) { return (bf16)f; }

// CDNA5 async copy: global -> LDS, 16 bytes per enabled lane (ASYNCcnt path).
__device__ inline void async_ld_b128(void* lds_ptr, const void* gptr) {
  unsigned       lds_off = (unsigned)(unsigned long long)lds_ptr;  // LDS offset
  unsigned long long ga  = (unsigned long long)gptr;
  asm volatile("global_load_async_to_lds_b128 %0, %1, off"
               :: "v"(lds_off), "v"(ga) : "memory");
}
__device__ inline void wait_asynccnt0() {
  asm volatile("s_wait_asynccnt 0x0" ::: "memory");
}

// A-matrix fragment 16x32 bf16: lane holds row=lane&15,
// K = kb..kb+15 contiguous with kb = (lane>>4)*16.
__device__ inline v16bf frag_a(const bf16* __restrict__ t, int ld, int lane) {
  const int row = lane & 15;
  const int kb  = (lane >> 4) << 4;
  v8bf lo = *(const v8bf*)(t + row * ld + kb);
  v8bf hi = *(const v8bf*)(t + row * ld + kb + 8);
  v16bf r;
#pragma unroll
  for (int i = 0; i < 8; i++) { r[i] = lo[i]; r[i + 8] = hi[i]; }
  return r;
}

// B-matrix fragment 32x16 bf16: lane = column n (mod 16), halves split K:
// lanes 0-15 hold K=0..15, lanes 16-31 hold K=16..31 — 16 contiguous bf16.
__device__ inline v16bf frag_b_lds(const bf16* __restrict__ t, int ld, int lane) {
  const int n  = lane & 15;
  const int kb = (lane >> 4) << 4;
  v8bf lo = *(const v8bf*)(t + n * ld + kb);
  v8bf hi = *(const v8bf*)(t + n * ld + kb + 8);
  v16bf r;
#pragma unroll
  for (int i = 0; i < 8; i++) { r[i] = lo[i]; r[i + 8] = hi[i]; }
  return r;
}

// B fragment whose 16 contiguous bf16 for this lane start at p.
__device__ inline v16bf frag_b_glob(const bf16* __restrict__ p) {
  v8bf lo = *(const v8bf*)p;
  v8bf hi = *(const v8bf*)(p + 8);
  v16bf r;
#pragma unroll
  for (int i = 0; i < 8; i++) { r[i] = lo[i]; r[i + 8] = hi[i]; }
  return r;
}

// ---------------------------------------------------------------------------
// Kernel 1: fused QKV projection.  out = x @ W^T + b  (bias is fp32).
// blockIdx.z selects Q/K/V.  Q,K -> (B,H,L,DH) bf16;  V -> (B,H,DH,L) bf16.
// 128x128 block tile, K-step 32, 8 waves each computing 32x64.
// ---------------------------------------------------------------------------
__global__ void __launch_bounds__(256)
qkv_gemm_kernel(const float* __restrict__ x,
                const float* __restrict__ Wq, const float* __restrict__ bq,
                const float* __restrict__ Wk, const float* __restrict__ bk,
                const float* __restrict__ Wv, const float* __restrict__ bv,
                bf16* __restrict__ qOut, bf16* __restrict__ kOut,
                bf16* __restrict__ vtOut) {
  __shared__ __align__(16) bf16 As[128 * 32];
  __shared__ __align__(16) bf16 Bs[128 * 32];

  const int tid  = threadIdx.x;
  const int lane = tid & 31;
  const int wave = tid >> 5;
  const int wm = wave & 3;      // 4 waves along M
  const int wn = wave >> 2;     // 2 waves along N
  const int bm = blockIdx.y * 128;
  const int bn = blockIdx.x * 128;
  const int which = blockIdx.z;

  const float* W    = (which == 0) ? Wq : (which == 1) ? Wk : Wv;
  const float* bias = (which == 0) ? bq : (which == 1) ? bk : bv;

  v8f acc[2][4];
#pragma unroll
  for (int i = 0; i < 2; i++)
#pragma unroll
    for (int j = 0; j < 4; j++) acc[i][j] = zero8();

  const int arow = tid >> 1;           // 128 rows, 2 threads per row
  const int acol = (tid & 1) << 4;     // 16 elements each

  for (int kt = 0; kt < DM; kt += 32) {
    // stage x-tile and W-tile as bf16 (conversion path must go via VGPRs)
    {
      const float4* gp = (const float4*)(x + (size_t)(bm + arow) * DM + kt + acol);
      bf16* dp = As + arow * 32 + acol;
      const float4* wp = (const float4*)(W + (size_t)(bn + arow) * DM + kt + acol);
      bf16* dq = Bs + arow * 32 + acol;
#pragma unroll
      for (int v = 0; v < 4; v++) {
        float4 f = gp[v];
        dp[v * 4 + 0] = f2bf(f.x); dp[v * 4 + 1] = f2bf(f.y);
        dp[v * 4 + 2] = f2bf(f.z); dp[v * 4 + 3] = f2bf(f.w);
        float4 g = wp[v];
        dq[v * 4 + 0] = f2bf(g.x); dq[v * 4 + 1] = f2bf(g.y);
        dq[v * 4 + 2] = f2bf(g.z); dq[v * 4 + 3] = f2bf(g.w);
      }
    }
    if (kt + 32 < DM) {
      __builtin_prefetch(x + (size_t)(bm + arow) * DM + kt + 32 + acol, 0, 1);
      __builtin_prefetch(W + (size_t)(bn + arow) * DM + kt + 32 + acol, 0, 1);
    }
    __syncthreads();

    v16bf a0 = frag_a(As + (wm * 32 +  0) * 32, 32, lane);
    v16bf a1 = frag_a(As + (wm * 32 + 16) * 32, 32, lane);
    v16bf b0 = frag_b_lds(Bs + (wn * 64 +  0) * 32, 32, lane);
    v16bf b1 = frag_b_lds(Bs + (wn * 64 + 16) * 32, 32, lane);
    v16bf b2 = frag_b_lds(Bs + (wn * 64 + 32) * 32, 32, lane);
    v16bf b3 = frag_b_lds(Bs + (wn * 64 + 48) * 32, 32, lane);

    acc[0][0] = WMMA_BF16(a0, b0, acc[0][0]);
    acc[0][1] = WMMA_BF16(a0, b1, acc[0][1]);
    acc[0][2] = WMMA_BF16(a0, b2, acc[0][2]);
    acc[0][3] = WMMA_BF16(a0, b3, acc[0][3]);
    acc[1][0] = WMMA_BF16(a1, b0, acc[1][0]);
    acc[1][1] = WMMA_BF16(a1, b1, acc[1][1]);
    acc[1][2] = WMMA_BF16(a1, b2, acc[1][2]);
    acc[1][3] = WMMA_BF16(a1, b3, acc[1][3]);
    __syncthreads();
  }

  // epilogue: C/D layout — VGPR r holds M = r + 8*(lane>>4), N = lane&15
  const int roff = (lane >> 4) << 3;
  const int ncol = lane & 15;
#pragma unroll
  for (int ms = 0; ms < 2; ms++)
#pragma unroll
    for (int ns = 0; ns < 4; ns++)
#pragma unroll
      for (int r = 0; r < 8; r++) {
        const int m = bm + wm * 32 + ms * 16 + roff + r;   // row in B*L
        const int n = bn + wn * 64 + ns * 16 + ncol;       // col in d_model
        const int b = m / L_, l = m % L_;
        const int h = n >> 6, dc = n & 63;
        const bf16 val = f2bf(acc[ms][ns][r] + bias[n]);
        const size_t bh = (size_t)b * H_ + h;
        if (which == 0)      qOut [(bh * L_ + l) * DH + dc] = val;
        else if (which == 1) kOut [(bh * L_ + l) * DH + dc] = val;
        else                 vtOut[(bh * DH + dc) * L_ + l] = val;
      }
}

// ---------------------------------------------------------------------------
// Kernel 2: ProbSparse attention for one (b, h, 16-query tile).
// Scores via WMMA -> f32 LDS; exact U-th-largest threshold via register-cached
// 32-bit radix select; masked softmax -> bf16 probs; context via WMMA +
// ds_add_f32.  Q tile staged with global_load_async_to_lds_b128 (ASYNCcnt).
// ---------------------------------------------------------------------------
__global__ void __launch_bounds__(256)
probsparse_attn_kernel(const bf16* __restrict__ qW, const bf16* __restrict__ kW,
                       const bf16* __restrict__ vtW, bf16* __restrict__ ctxW) {
  extern __shared__ char smem[];
  float* Sc  = (float*)smem;                                   // 16 x 2048 f32
  bf16*  Pb  = (bf16*)(smem + 16 * 2048 * 4);                  // 16 x 2048 bf16
  bf16*  Qs  = (bf16*)(smem + 16 * 2048 * 4 + 16 * 2048 * 2);  // 16 x 64 bf16
  float* Ctx = (float*)(smem + 16 * 2048 * 4 + 16 * 2048 * 2 + 16 * 64 * 2); // 16x64

  const int tid  = threadIdx.x;
  const int lane = tid & 31;
  const int wave = tid >> 5;
  const int qb = blockIdx.x * 16;
  const size_t bh = (size_t)blockIdx.z * H_ + blockIdx.y;

  const bf16* Qg = qW + (bh * L_ + qb) * DH;   // 16 contiguous rows of 64
  const bf16* Kg = kW + bh * L_ * DH;
  const bf16* Vt = vtW + bh * DH * L_;

  // Q tile: 2KB raw bf16 copy -> async DMA to LDS, 128 lanes x 16B
  if (tid < 128) async_ld_b128(Qs + tid * 8, Qg + tid * 8);
  for (int i = tid; i < 16 * DH; i += 256) Ctx[i] = 0.f;
  wait_asynccnt0();
  __syncthreads();

  // ---- phase A: scores S = (Q K^T) / 8.  wave w owns key cols [w*256, +256)
  {
    v16bf a0 = frag_a(Qs +  0, DH, lane);
    v16bf a1 = frag_a(Qs + 32, DH, lane);
    const int nB   = lane & 15;
    const int kOff = (lane >> 4) << 4;
    const int roff = (lane >> 4) << 3;
    for (int nt = 0; nt < 16; nt++) {
      const int kb = wave * 256 + nt * 16;
      v8f acc = zero8();
      v16bf b0 = frag_b_glob(Kg + (size_t)(kb + nB) * DH +  0 + kOff);
      v16bf b1 = frag_b_glob(Kg + (size_t)(kb + nB) * DH + 32 + kOff);
      acc = WMMA_BF16(a0, b0, acc);
      acc = WMMA_BF16(a1, b1, acc);
#pragma unroll
      for (int r = 0; r < 8; r++)
        Sc[(roff + r) * 2048 + kb + nB] = acc[r] * 0.125f;   // 1/sqrt(64)
    }
  }
  __syncthreads();

  // ---- phase B: per-row exact top-U threshold + masked softmax (wave/row).
  // Scores cached in registers: 64 values + 64 sortable keys per lane.
  for (int rr = 0; rr < 2; rr++) {
    const int row = wave * 2 + rr;
    const float* S = Sc + row * 2048;
    float    vv[64];
    unsigned kk[64];
#pragma unroll
    for (int i = 0; i < 64; i++) {
      const float f = S[lane + 32 * i];
      vv[i] = f;
      const unsigned u = __float_as_uint(f);
      kk[i] = (u & 0x80000000u) ? ~u : (u | 0x80000000u);
    }
    // radix select: after the loop, prefix == key of the U-th largest score
    unsigned prefix = 0u;
    for (int bit = 31; bit >= 0; bit--) {
      const unsigned t = prefix | (1u << bit);
      int cnt = 0;
#pragma unroll
      for (int i = 0; i < 64; i++) cnt += (kk[i] >= t) ? 1 : 0;
#pragma unroll
      for (int off = 16; off; off >>= 1) cnt += __shfl_xor(cnt, off, 32);
      if (cnt >= U_) prefix = t;
    }
    // row max (max of kept == overall max)
    float m = -3.402823466e38f;
#pragma unroll
    for (int i = 0; i < 64; i++) m = fmaxf(m, vv[i]);
#pragma unroll
    for (int off = 16; off; off >>= 1) m = fmaxf(m, __shfl_xor(m, off, 32));
    // fused mask + exp (one v_exp_f32 per element), then sum
    float s = 0.f;
#pragma unroll
    for (int i = 0; i < 64; i++) {
      vv[i] = (kk[i] >= prefix) ? __expf(vv[i] - m) : 0.f;
      s += vv[i];
    }
#pragma unroll
    for (int off = 16; off; off >>= 1) s += __shfl_xor(s, off, 32);
    const float inv = 1.f / s;

    bf16* P = Pb + row * 2048;
#pragma unroll
    for (int i = 0; i < 64; i++) P[lane + 32 * i] = f2bf(vv[i] * inv);
  }
  __syncthreads();

  // ---- phase C: context = P @ V.  wave w reduces its 256-key slice.
  {
    v8f acc[4];
#pragma unroll
    for (int j = 0; j < 4; j++) acc[j] = zero8();
    const int nB   = lane & 15;
    const int kOff = (lane >> 4) << 4;
    for (int kc = 0; kc < 8; kc++) {
      const int kb = wave * 256 + kc * 32;
      v16bf a = frag_a(Pb + kb, 2048, lane);
#pragma unroll
      for (int nt = 0; nt < 4; nt++) {
        v16bf bb = frag_b_glob(Vt + (size_t)(nt * 16 + nB) * L_ + kb + kOff);
        acc[nt] = WMMA_BF16(a, bb, acc[nt]);
      }
    }
    const int roff = (lane >> 4) << 3;
#pragma unroll
    for (int nt = 0; nt < 4; nt++)
#pragma unroll
      for (int r = 0; r < 8; r++)
        atomicAdd(&Ctx[(roff + r) * DH + nt * 16 + nB], acc[nt][r]);  // ds_add_f32
  }
  __syncthreads();

  // context -> (B, L, d_model) bf16
  bf16* C = ctxW + ((size_t)blockIdx.z * L_ + qb) * DM + blockIdx.y * DH;
  for (int i = tid; i < 16 * DH; i += 256) {
    const int r = i >> 6, c = i & 63;
    C[(size_t)r * DM + c] = f2bf(Ctx[i]);
  }
}

// ---------------------------------------------------------------------------
// Kernel 3: output projection.  out = ctx(bf16) @ Wo^T + bo  (fp32 out)
// A-tile is already bf16 -> staged with async DMA to LDS.
// ---------------------------------------------------------------------------
__global__ void __launch_bounds__(256)
oproj_gemm_kernel(const bf16* __restrict__ ctx, const float* __restrict__ Wo,
                  const float* __restrict__ bo, float* __restrict__ out) {
  __shared__ __align__(16) bf16 As[128 * 32];
  __shared__ __align__(16) bf16 Bs[128 * 32];

  const int tid  = threadIdx.x;
  const int lane = tid & 31;
  const int wave = tid >> 5;
  const int wm = wave & 3, wn = wave >> 2;
  const int bm = blockIdx.y * 128;
  const int bn = blockIdx.x * 128;

  v8f acc[2][4];
#pragma unroll
  for (int i = 0; i < 2; i++)
#pragma unroll
    for (int j = 0; j < 4; j++) acc[i][j] = zero8();

  const int arow = tid >> 1;
  const int acol = (tid & 1) << 4;

  for (int kt = 0; kt < DM; kt += 32) {
    {
      // A (bf16 raw): async global->LDS DMA, 2 x 16B per thread
      const bf16* ga = ctx + (size_t)(bm + arow) * DM + kt + acol;
      async_ld_b128(As + arow * 32 + acol,     ga);
      async_ld_b128(As + arow * 32 + acol + 8, ga + 8);
      // B (fp32 -> bf16): conversion path via VGPRs
      const float4* wp = (const float4*)(Wo + (size_t)(bn + arow) * DM + kt + acol);
      bf16* dq = Bs + arow * 32 + acol;
#pragma unroll
      for (int v = 0; v < 4; v++) {
        float4 g = wp[v];
        dq[v * 4 + 0] = f2bf(g.x); dq[v * 4 + 1] = f2bf(g.y);
        dq[v * 4 + 2] = f2bf(g.z); dq[v * 4 + 3] = f2bf(g.w);
      }
    }
    if (kt + 32 < DM)
      __builtin_prefetch(Wo + (size_t)(bn + arow) * DM + kt + 32 + acol, 0, 1);
    wait_asynccnt0();
    __syncthreads();

    v16bf a0 = frag_a(As + (wm * 32 +  0) * 32, 32, lane);
    v16bf a1 = frag_a(As + (wm * 32 + 16) * 32, 32, lane);
    v16bf b0 = frag_b_lds(Bs + (wn * 64 +  0) * 32, 32, lane);
    v16bf b1 = frag_b_lds(Bs + (wn * 64 + 16) * 32, 32, lane);
    v16bf b2 = frag_b_lds(Bs + (wn * 64 + 32) * 32, 32, lane);
    v16bf b3 = frag_b_lds(Bs + (wn * 64 + 48) * 32, 32, lane);

    acc[0][0] = WMMA_BF16(a0, b0, acc[0][0]);
    acc[0][1] = WMMA_BF16(a0, b1, acc[0][1]);
    acc[0][2] = WMMA_BF16(a0, b2, acc[0][2]);
    acc[0][3] = WMMA_BF16(a0, b3, acc[0][3]);
    acc[1][0] = WMMA_BF16(a1, b0, acc[1][0]);
    acc[1][1] = WMMA_BF16(a1, b1, acc[1][1]);
    acc[1][2] = WMMA_BF16(a1, b2, acc[1][2]);
    acc[1][3] = WMMA_BF16(a1, b3, acc[1][3]);
    __syncthreads();
  }

  const int roff = (lane >> 4) << 3;
  const int ncol = lane & 15;
#pragma unroll
  for (int ms = 0; ms < 2; ms++)
#pragma unroll
    for (int ns = 0; ns < 4; ns++)
#pragma unroll
      for (int r = 0; r < 8; r++) {
        const int m = bm + wm * 32 + ms * 16 + roff + r;
        const int n = bn + wn * 64 + ns * 16 + ncol;
        out[(size_t)m * DM + n] = acc[ms][ns][r] + bo[n];
      }
}

// ---------------------------------------------------------------------------
extern "C" void kernel_launch(void* const* d_in, const int* in_sizes, int n_in,
                              void* d_out, int out_size, void* d_ws, size_t ws_size,
                              hipStream_t stream) {
  const float* x  = (const float*)d_in[0];
  const float* Wq = (const float*)d_in[1];
  const float* bq = (const float*)d_in[2];
  const float* Wk = (const float*)d_in[3];
  const float* bk = (const float*)d_in[4];
  const float* Wv = (const float*)d_in[5];
  const float* bv = (const float*)d_in[6];
  const float* Wo = (const float*)d_in[7];
  const float* bo = (const float*)d_in[8];

  const size_t perTensor = (size_t)B_ * H_ * L_ * DH;  // 4M bf16 elems
  bf16* qW   = (bf16*)d_ws;
  bf16* kW   = qW + perTensor;
  bf16* vtW  = kW + perTensor;
  bf16* ctxW = vtW + perTensor;

  dim3 blk(256);
  qkv_gemm_kernel<<<dim3(DM / 128, (B_ * L_) / 128, 3), blk, 0, stream>>>(
      x, Wq, bq, Wk, bk, Wv, bv, qW, kW, vtW);

  const size_t smem = 16 * 2048 * 4 + 16 * 2048 * 2 + 16 * 64 * 2 + 16 * 64 * 4;
  probsparse_attn_kernel<<<dim3(L_ / 16, H_, B_), blk, smem, stream>>>(
      qW, kW, vtW, ctxW);

  oproj_gemm_kernel<<<dim3(DM / 128, (B_ * L_) / 128), blk, 0, stream>>>(
      ctxW, Wo, bo, (float*)d_out);
}
